// HeteroMultiHeadAttention_20177756356842
// MI455X (gfx1250) — matrix operand order repaired and checked by reading, not compiled, not run on hardware
//
#include <hip/hip_runtime.h>
#include <hip/hip_bf16.h>
#include <math.h>

// ---------------------------------------------------------------------------
// HeteroMultiHeadAttention (GAT-style) for MI455X / gfx1250, wave32 + WMMA
// + async global->LDS staging + non-temporal attn stream-out.
//
// B=8, Ns=Nt=1024, E=256, NH=8, Dh=32, INNER=256, NEG_SLOPE=0.2
//
// d_out (fp32): [0..2097152) output (B,Ns,256) | then attn^T (B,NH,Ns,Nt)
// ---------------------------------------------------------------------------

#define B_     8
#define N_     1024
#define E_     256
#define NH_    8
#define DH_    32
#define INNER_ 256
#define SLOPE_ 0.2f
#define NEGINF (-__builtin_inff())

typedef __attribute__((ext_vector_type(16))) __bf16 v16bf;
typedef __attribute__((ext_vector_type(8)))  float  v8f;
typedef __attribute__((ext_vector_type(4)))  float  f32x4;

// ---------- WMMA tile loaders (layouts per CDNA5 ISA §7.12.2, wave32) -------
// 16-bit A 16x32 (MxK): lanes 0-15 hold row M=lane, elems [0..7]=K0..7,
// [8..15]=K16..23; lanes 16-31 hold row M=lane-16, [0..7]=K8..15, [8..15]=K24..31.
__device__ inline v16bf load_A_f32(const float* __restrict__ base, int ld,
                                   int row0, int k0, int lane) {
    const int r  = lane & 15;
    const int kk = k0 + ((lane >> 4) << 3);
    const float* p = base + (size_t)(row0 + r) * ld + kk;
    v16bf a;
#pragma unroll
    for (int i = 0; i < 8; ++i) { a[i] = (__bf16)p[i]; a[8 + i] = (__bf16)p[16 + i]; }
    return a;
}

__device__ inline v16bf load_A_bf16(const __bf16* __restrict__ base, int ld,
                                    int row0, int k0, int lane) {
    const int r  = lane & 15;
    const int kk = k0 + ((lane >> 4) << 3);
    const __bf16* p = base + (size_t)(row0 + r) * ld + kk;
    v16bf a;
#pragma unroll
    for (int i = 0; i < 8; ++i) { a[i] = p[i]; a[8 + i] = p[16 + i]; }
    return a;
}

// 16-bit B 32x16 (KxN): lanes 0-15 hold column N=lane, K=0..7,16..23; lanes 16-31
// hold column N=lane-16, K=8..15,24..31. baseT is K-contiguous per output column
// (transposed weight / V^T); works for both global and LDS pointers.
__device__ inline v16bf load_B_T(const __bf16* baseT, int ld,
                                 int col0, int k0, int lane) {
    const int c  = lane & 15;
    const int kk = k0 + ((lane >> 4) << 3);
    const __bf16* p = baseT + (size_t)(col0 + c) * ld + kk;
    v16bf b;
#pragma unroll
    for (int i = 0; i < 8; ++i) { b[i] = p[i]; b[8 + i] = p[16 + i]; }
    return b;
}

__device__ inline v8f wmma_bf16(v16bf a, v16bf b, v8f c) {
    return __builtin_amdgcn_wmma_f32_16x16x32_bf16(false, a, false, b,
                                                   (short)0, c, false, false);
}

__device__ inline float leaky(float v) { return v >= 0.f ? v : SLOPE_ * v; }

// ---------------------------------------------------------------------------
// Kernel 0: fold attention vectors into the projections:
//   c_src[e][h] = sum_d W_src[e][h*32+d] * a_src[h][d]  => src_proj is never
// materialized; score_src is just src @ c_src. Same for trg. Also transpose+
// convert W_trg and W_out to bf16 (K-contiguous per output column).
// ---------------------------------------------------------------------------
__global__ void hmha_prep(const float* __restrict__ W_src,
                          const float* __restrict__ W_trg,
                          const float* __restrict__ a_src,
                          const float* __restrict__ a_trg,
                          const float* __restrict__ W_out,
                          float* __restrict__ c_src, float* __restrict__ c_trg,
                          __bf16* __restrict__ WtT, __bf16* __restrict__ WoT) {
    const int e = threadIdx.x;  // 256 threads, 1 block
#pragma unroll
    for (int h = 0; h < NH_; ++h) {
        float s1 = 0.f, s2 = 0.f;
        for (int d = 0; d < DH_; ++d) {
            s1 += W_src[e * INNER_ + h * DH_ + d] * a_src[h * DH_ + d];
            s2 += W_trg[e * INNER_ + h * DH_ + d] * a_trg[h * DH_ + d];
        }
        c_src[e * NH_ + h] = s1;
        c_trg[e * NH_ + h] = s2;
    }
    for (int i = e; i < E_ * INNER_; i += blockDim.x) {
        int ee = i >> 8, cc = i & 255;
        WtT[(size_t)cc * E_ + ee] = (__bf16)W_trg[i];
        WoT[(size_t)cc * E_ + ee] = (__bf16)W_out[i];
    }
}

// ---------------------------------------------------------------------------
// Kernel S: score_src[b,h,n] = src[b,n,:] . c_src[:,h]; same for trg. Tiny.
// ---------------------------------------------------------------------------
__global__ void hmha_scores(const float* __restrict__ src,
                            const float* __restrict__ trg,
                            const float* __restrict__ c_src,
                            const float* __restrict__ c_trg,
                            float* __restrict__ ssbuf,   // (B,NH,Ns)
                            float* __restrict__ stbuf) { // (B,NH,Nt)
    const int idx = blockIdx.x * blockDim.x + threadIdx.x;  // B*NH*N = 65536
    const int b = idx >> 13;
    const int h = (idx >> 10) & 7;
    const int n = idx & (N_ - 1);
    const float* srow = src + ((size_t)b * N_ + n) * E_;
    const float* trow = trg + ((size_t)b * N_ + n) * E_;
    float accs = 0.f, acct = 0.f;
    for (int e = 0; e < E_; ++e) {
        accs = fmaf(srow[e], c_src[e * NH_ + h], accs);
        acct = fmaf(trow[e], c_trg[e * NH_ + h], acct);
    }
    ssbuf[((size_t)b * NH_ + h) * N_ + n] = accs;
    stbuf[((size_t)b * NH_ + h) * N_ + n] = acct;
}

// ---------------------------------------------------------------------------
// Kernel P: trg_proj = trg @ W_trg via bf16 WMMA, stored as bf16 V^T with
// layout (B, NH, Dh, Nt) so attn@V B-tiles are contiguous. One wave = one
// 16x16 tile, K-loop 256/32 = 8 WMMAs; packed 16B stores (8 consecutive t).
// ---------------------------------------------------------------------------
__global__ void __launch_bounds__(256) hmha_proj(const float* __restrict__ trg,
                                                 const __bf16* __restrict__ WtT,
                                                 __bf16* __restrict__ vT) {
    const int lane = threadIdx.x & 31;
    const int wid  = blockIdx.x * (blockDim.x >> 5) + (threadIdx.x >> 5);
    const int colTile = wid & 15;          // 256/16 col tiles
    const int rowTile = (wid >> 4) & 63;   // 1024/16 row tiles
    const int b  = wid >> 10;              // 8 batches
    const int t0 = rowTile * 16;
    const int n0 = colTile * 16;
    const float* Abase = trg + (size_t)b * N_ * E_;

    v8f acc = {};
#pragma unroll
    for (int k0 = 0; k0 < E_; k0 += 32) {
        v16bf a  = load_A_f32(Abase, E_, t0, k0, lane);
        v16bf bb = load_B_T(WtT, E_, n0, k0, lane);
        acc = wmma_bf16(a, bb, acc);
    }
    // C layout: VGPR r = (M=r, N=lane) lanes<16; (M=r+8, N=lane-16) lanes>=16.
    // For a fixed lane the 8 values are 8 consecutive t -> one 16B store.
    const int c  = n0 + (lane & 15);
    const int h  = c >> 5, d = c & 31;
    const int hi = lane >> 4;
    __bf16* dst = vT + (((size_t)b * NH_ + h) * DH_ + d) * N_ + t0 + hi * 8;
    union { __bf16 hh[8]; f32x4 v; } u;
#pragma unroll
    for (int r = 0; r < 8; ++r) u.hh[r] = (__bf16)acc[r];
    *(f32x4*)dst = u.v;
}

// ---------------------------------------------------------------------------
// Kernel A: fused masked-LeakyReLU-softmax + attn@V. One workgroup per
// (b, h, 128 src rows); the (Dh x Nt) V^T slice (64KB) is staged into LDS
// once via CDNA5 async global->LDS (ASYNCcnt), the masked score row (4KB)
// is pre-folded into LDS, and the 256MB attn tensor is streamed to d_out
// with non-temporal stores (write-once, never re-read -> don't pollute L2).
// ---------------------------------------------------------------------------
__global__ void __launch_bounds__(256) hmha_attn(
        const float* __restrict__ ssbuf,
        const float* __restrict__ stbuf,
        const unsigned char* __restrict__ src_mask,
        const unsigned char* __restrict__ trg_mask,
        const __bf16* __restrict__ vT,
        float* __restrict__ out_attn,  // (B,NH,Ns,Nt)
        __bf16* __restrict__ agg) {    // (B,Ns,INNER)
    __shared__ float  stm[N_];          // masked score_trg row (-inf folded in)
    __shared__ __bf16 vsh[DH_ * N_];    // V^T slice [d][t], 64KB

    const int tid  = threadIdx.x;
    const int lane = tid & 31;
    const int wave = tid >> 5;
    const int b = blockIdx.z, h = blockIdx.y;

    const float* st = stbuf + ((size_t)b * NH_ + h) * N_;
    const unsigned char* tm = trg_mask + b * N_;
    const __bf16* vsrc = vT + ((size_t)b * NH_ + h) * DH_ * N_;

    // ---- stage masked scores into LDS ----
    for (int t = tid; t < N_; t += 256)
        stm[t] = tm[t] ? st[t] : NEGINF;

    // ---- stage V^T slice via async global->LDS b128 copies ----
    {
        const unsigned lds0 = (unsigned)(size_t)(void*)vsh;
        const char*    g0   = (const char*)vsrc;
#pragma unroll
        for (int i = 0; i < (DH_ * N_ * 2) / (256 * 16); ++i) {  // 16 iters
            const unsigned ldsa = lds0 + (unsigned)((i * 256 + tid) * 16);
            const char*    ga   = g0 + (size_t)(i * 256 + tid) * 16;
            asm volatile("global_load_async_to_lds_b128 %0, %1, off"
                         :: "v"(ldsa), "v"(ga) : "memory");
        }
        asm volatile("s_wait_asynccnt 0x0" ::: "memory");
    }
    __syncthreads();

    const int s0 = (blockIdx.x * (blockDim.x >> 5) + wave) * 16;
    const int r  = lane & 15;
    const int hi = lane >> 4;
    const int s  = s0 + r;

    float ss = ssbuf[((size_t)b * NH_ + h) * N_ + s];
    if (!src_mask[b * N_ + s]) ss = NEGINF;  // whole row masked -> attn row = 0

    // pass 1: row max (lane pair (r, r+16) splits the t range)
    float m = NEGINF;
    for (int i = 0; i < N_ / 2; ++i)
        m = fmaxf(m, leaky(ss + stm[hi * (N_ / 2) + i]));
    m = fmaxf(m, __shfl_xor(m, 16, 32));

    // pass 2: sum of exp (masked t: v=-inf -> exp(-inf)=0; fully-masked row
    // guarded by finite-m check, reproducing nan_to_num(attn)=0)
    float sum = 0.f;
    if (m > -3.0e38f) {
        for (int i = 0; i < N_ / 2; ++i)
            sum += __expf(leaky(ss + stm[hi * (N_ / 2) + i]) - m);
    }
    sum += __shfl_xor(sum, 16, 32);
    const float inv  = sum > 0.f ? 1.f / sum : 0.f;
    const bool  live = (m > -3.0e38f);

    // pass 3: build P tiles in WMMA A layout, stream attn (NT), accumulate P@V
    v8f acc0 = {}, acc1 = {};
    float* arow = out_attn + (((size_t)b * NH_ + h) * N_ + s) * (size_t)N_;

    for (int tb = 0; tb < N_; tb += 32) {
        float p[16];
#pragma unroll
        for (int j = 0; j < 16; ++j) {
            const int k = (j < 8 ? j : j + 8) + hi * 8;  // A-layout K offset
            p[j] = live ? __expf(leaky(ss + stm[tb + k]) - m) * inv : 0.f;
        }
        // fp32 attn out: four contiguous 16B groups per lane, non-temporal
#pragma unroll
        for (int g = 0; g < 4; ++g) {
            const int j = g * 4;
            const int k = (j < 8 ? j : j + 8) + hi * 8;
            f32x4 v4 = {p[j], p[j + 1], p[j + 2], p[j + 3]};
            __builtin_nontemporal_store(v4, (f32x4*)(arow + tb + k));
        }
        v16bf a;
#pragma unroll
        for (int j = 0; j < 16; ++j) a[j] = (__bf16)p[j];

        v16bf b0 = load_B_T(vsh, N_, 0,  tb, lane);  // ds_load: V cols d=0..15
        v16bf b1 = load_B_T(vsh, N_, 16, tb, lane);  // ds_load: V cols d=16..31
        acc0 = wmma_bf16(a, b0, acc0);
        acc1 = wmma_bf16(a, b1, acc1);
    }

    // aggregated tile (C layout) -> bf16 (B, Ns, INNER)
    const int n = lane & 15;
#pragma unroll
    for (int rr = 0; rr < 8; ++rr) {
        const int srow = s0 + rr + hi * 8;
        __bf16* dst = agg + ((size_t)b * N_ + srow) * INNER_ + h * DH_;
        dst[n]      = (__bf16)acc0[rr];
        dst[16 + n] = (__bf16)acc1[rr];
    }
}

// ---------------------------------------------------------------------------
// Kernel O: output = aggregated(bf16) @ W_out via WMMA -> fp32 d_out.
// ---------------------------------------------------------------------------
__global__ void __launch_bounds__(256) hmha_out(const __bf16* __restrict__ agg,
                                                const __bf16* __restrict__ WoT,
                                                float* __restrict__ out) {
    const int lane = threadIdx.x & 31;
    const int wid  = blockIdx.x * (blockDim.x >> 5) + (threadIdx.x >> 5);
    const int colTile = wid & 15;
    const int rowTile = (wid >> 4) & 63;
    const int b  = wid >> 10;
    const int s0 = rowTile * 16;
    const int n0 = colTile * 16;
    const __bf16* Abase = agg + (size_t)b * N_ * INNER_;

    v8f acc = {};
#pragma unroll
    for (int k0 = 0; k0 < INNER_; k0 += 32) {
        v16bf a  = load_A_bf16(Abase, INNER_, s0, k0, lane);
        v16bf bb = load_B_T(WoT, INNER_, n0, k0, lane);
        acc = wmma_bf16(a, bb, acc);
    }
    const int c  = n0 + (lane & 15);
    const int hi = lane >> 4;
#pragma unroll
    for (int r = 0; r < 8; ++r)
        out[((size_t)b * N_ + s0 + r + hi * 8) * E_ + c] = acc[r];
}

// ---------------------------------------------------------------------------
extern "C" void kernel_launch(void* const* d_in, const int* in_sizes, int n_in,
                              void* d_out, int out_size, void* d_ws, size_t ws_size,
                              hipStream_t stream) {
    const float* src  = (const float*)d_in[0];
    const float* trg  = (const float*)d_in[1];
    const unsigned char* src_mask = (const unsigned char*)d_in[2];
    const unsigned char* trg_mask = (const unsigned char*)d_in[3];
    const float* W_src = (const float*)d_in[4];
    const float* W_trg = (const float*)d_in[5];
    const float* a_src = (const float*)d_in[6];
    const float* a_trg = (const float*)d_in[7];
    const float* W_out = (const float*)d_in[8];

    float* out      = (float*)d_out;                          // (B,Ns,E)
    float* out_attn = (float*)d_out + (size_t)B_ * N_ * E_;   // (B,NH,Ns,Nt)

    // workspace carve-up (256B aligned)
    char* ws = (char*)d_ws;
    size_t off = 0;
    auto take = [&](size_t bytes) { char* p = ws + off; off = (off + bytes + 255) & ~(size_t)255; return p; };
    float*  c_src = (float*)take(E_ * NH_ * sizeof(float));
    float*  c_trg = (float*)take(E_ * NH_ * sizeof(float));
    __bf16* WtT   = (__bf16*)take((size_t)E_ * INNER_ * sizeof(__bf16));
    __bf16* WoT   = (__bf16*)take((size_t)E_ * INNER_ * sizeof(__bf16));
    float*  ssbuf = (float*)take((size_t)B_ * NH_ * N_ * sizeof(float));
    float*  stbuf = (float*)take((size_t)B_ * NH_ * N_ * sizeof(float));
    __bf16* vT    = (__bf16*)take((size_t)B_ * NH_ * DH_ * N_ * sizeof(__bf16));
    __bf16* agg   = (__bf16*)take((size_t)B_ * N_ * INNER_ * sizeof(__bf16));
    (void)ws_size; (void)in_sizes; (void)n_in; (void)out_size;

    hmha_prep<<<1, 256, 0, stream>>>(W_src, W_trg, a_src, a_trg, W_out,
                                     c_src, c_trg, WtT, WoT);
    hmha_scores<<<(B_ * NH_ * N_) / 256, 256, 0, stream>>>(src, trg, c_src, c_trg,
                                                           ssbuf, stbuf);
    hmha_proj<<<(B_ * (N_ / 16) * (INNER_ / 16)) / 8, 256, 0, stream>>>(trg, WtT, vT);
    hmha_attn<<<dim3(N_ / (16 * 8), NH_, B_), 256, 0, stream>>>(ssbuf, stbuf,
                                                                src_mask, trg_mask,
                                                                vT, out_attn, agg);
    hmha_out<<<(B_ * (N_ / 16) * (E_ / 16)) / 8, 256, 0, stream>>>(agg, WoT, out);
}
